// XCA_26903675142844
// MI455X (gfx1250) — matrix operand order repaired and checked by reading, compiled, not run on hardware
//
#include <hip/hip_runtime.h>
#include <cstdint>

// XCA (cross-covariance attention) for MI455X / gfx1250.
// B=4, S=4096, HID=1024, H=16, Dh=64. bf16 WMMA (16x16x32) with fp32 accum;
// GEMMs stage block tiles into LDS via the Tensor Data Mover (tensor_load_to_lds).

typedef __attribute__((ext_vector_type(16))) __bf16 v16bf;
typedef __attribute__((ext_vector_type(8)))  __bf16 v8bf;
typedef __attribute__((ext_vector_type(8)))  float  v8f;
typedef __attribute__((ext_vector_type(4)))  unsigned int v4u;
typedef __attribute__((ext_vector_type(8)))  int    v8i;
typedef __attribute__((ext_vector_type(4)))  int    v4i;

#define B_   4
#define S_   4096
#define HID_ 1024
#define H_   16
#define DH_  64

#define LDS_STRIDE 40   // bf16 elems per staged row: 64B data + 16B pad (bank-conflict-free)

// ---------------- WMMA fragment helpers (wave32 layouts, CDNA5 ISA) ----

// A-matrix 16x32 bf16: lane l holds row M=l%16; elems 0..15 -> K=hf*16+0..15
// (hf = lane/16).
__device__ __forceinline__ v16bf load_frag_a(const __bf16* base, int ld, int l16, int hf) {
    const __bf16* p = base + (size_t)l16 * ld + hf * 16;
    v8bf lo = *(const v8bf*)(p);
    v8bf hi = *(const v8bf*)(p + 8);
    v16bf a;
#pragma unroll
    for (int e = 0; e < 8; ++e) { a[e] = lo[e]; a[e + 8] = hi[e]; }
    return a;
}

// B-matrix 32x16 bf16: lane l holds column N=l%16; 16 contiguous K at hf*16.
// Memory holds B[k][n] = W[n][k], so per lane this is one contiguous 32B load.
__device__ __forceinline__ v16bf load_frag_b(const __bf16* base, int ld, int l16, int hf) {
    const __bf16* p = base + (size_t)l16 * ld + hf * 16;
    return *(const v16bf*)(p);
}

__device__ __forceinline__ v8f wmma_bf16(v16bf a, v16bf b, v8f c) {
    return __builtin_amdgcn_wmma_f32_16x16x32_bf16(
        false, a, false, b, (short)0, c, false, false);
}

// ---------------- Tensor Data Mover: 2D bf16 tile -> LDS (CDNA5 ISA) --
// Loads tile (tile0 x tile1 rows) of a row-major bf16 tensor into LDS with 16B of
// padding appended per 64B row (pad_interval=3 -> 64B, pad_amount=3 -> 4 DWORDs),
// producing rows of LDS_STRIDE=40 bf16. 6-arg builtin form (clang-23/therock).

__device__ __forceinline__ void tdm_load_tile_2d(const void* gaddr, unsigned lds_off,
                                                 unsigned dim0, unsigned dim1,
                                                 unsigned tile0, unsigned tile1,
                                                 unsigned long long stride0) {
    unsigned long long ga = (unsigned long long)(uintptr_t)gaddr;
    v4u g0;
    g0[0] = 1u;                                        // count=1 (valid), user mode
    g0[1] = lds_off;                                   // lds_addr (bytes)
    g0[2] = (unsigned)(ga & 0xFFFFFFFFu);              // global_addr[31:0]
    g0[3] = (unsigned)((ga >> 32) & 0x1FFFFFFu)        // global_addr[56:32]
          | (2u << 30);                                // type=2 ("image")
    v8i g1;
    g1[0] = (int)((1u << 16)                           // data_size=1 (2 bytes)
               | (1u << 20)                            // pad_enable
               | (3u << 22)                            // pad_interval: 16 DW = 64B
               | (3u << 25));                          // pad_amount:   4 DW = 16B
    g1[1] = (int)((dim0 & 0xFFFFu) << 16);             // tensor_dim0[15:0]
    g1[2] = (int)(((dim0 >> 16) & 0xFFFFu) | ((dim1 & 0xFFFFu) << 16));
    g1[3] = (int)(((dim1 >> 16) & 0xFFFFu) | ((tile0 & 0xFFFFu) << 16));
    g1[4] = (int)(tile1 & 0xFFFFu);                    // tile_dim1 (tile_dim2=0)
    g1[5] = (int)(stride0 & 0xFFFFFFFFull);            // tensor_dim0_stride[31:0]
    g1[6] = (int)((stride0 >> 32) & 0xFFFFull);        // stride[47:32], dim1_stride=0
    g1[7] = 0;
    v4i z4; z4[0] = 0; z4[1] = 0; z4[2] = 0; z4[3] = 0;
    v8i z8;
#pragma unroll
    for (int i = 0; i < 8; ++i) z8[i] = 0;
    __builtin_amdgcn_tensor_load_to_lds(g0, g1, z4, z4, z8, 0);
}

__device__ __forceinline__ unsigned lds_offset_of(const void* p) {
    // Generic LDS address = {aperture_hi, offset[31:0]} -> low 32 bits are LDS bytes.
    return (unsigned)(uintptr_t)p;
}

// ---------------- K0: fp32 -> bf16 conversion ------------------------------------

__global__ void k_to_bf16(const float* __restrict__ src, __bf16* __restrict__ dst, int n) {
    int i = blockIdx.x * blockDim.x + threadIdx.x;
    if (i < n) dst[i] = (__bf16)src[i];
}

// ---------------- K1: QKV GEMM, TDM-staged, scatter to [3,B,H,Dh,S] fp32 ---------
// Block tile 128(M) x 128(N), K-step 32; wave computes 32x64 (2x4 tiles).

__global__ void k_qkv_gemm(const __bf16* __restrict__ X, const __bf16* __restrict__ W,
                           const float* __restrict__ bias, float* __restrict__ qkvf) {
    __shared__ __bf16 As[128][LDS_STRIDE];
    __shared__ __bf16 Bs[128][LDS_STRIDE];

    int bm = blockIdx.x & 127;        // 128 M-blocks
    int bn = blockIdx.x >> 7;         // 24  N-blocks
    int mbase = bm * 128, nbase = bn * 128;
    int wave = threadIdx.x >> 5, lane = threadIdx.x & 31;
    int l16 = lane & 15, hf = lane >> 4;
    int msub = wave & 3;              // 4 x 32 rows
    int nsub = wave >> 2;             // 2 x 64 cols

    unsigned ldsA = lds_offset_of(&As[0][0]);
    unsigned ldsB = lds_offset_of(&Bs[0][0]);

    v8f acc[2][4] = {};
    for (int k0 = 0; k0 < HID_; k0 += 32) {
        if (wave == 0) {
            tdm_load_tile_2d(X + (size_t)mbase * HID_ + k0, ldsA,
                             HID_, 16384u, 32u, 128u, HID_);
            tdm_load_tile_2d(W + (size_t)nbase * HID_ + k0, ldsB,
                             HID_, 3072u, 32u, 128u, HID_);
            __builtin_amdgcn_s_wait_tensorcnt((short)0);
        }
        __syncthreads();
        v16bf a0 = load_frag_a(&As[msub * 32][0],      LDS_STRIDE, l16, hf);
        v16bf a1 = load_frag_a(&As[msub * 32 + 16][0], LDS_STRIDE, l16, hf);
#pragma unroll
        for (int j = 0; j < 4; ++j) {
            v16bf bf = load_frag_b(&Bs[nsub * 64 + j * 16][0], LDS_STRIDE, l16, hf);
            acc[0][j] = wmma_bf16(a0, bf, acc[0][j]);
            acc[1][j] = wmma_bf16(a1, bf, acc[1][j]);
        }
        __syncthreads();
    }

    // Scatter: lane column n -> (comp,h,dh); 8 rows m -> contiguous s.
#pragma unroll
    for (int i = 0; i < 2; ++i) {
        int m0 = mbase + msub * 32 + i * 16 + hf * 8;
        int b_ = m0 >> 12, s0 = m0 & (S_ - 1);
#pragma unroll
        for (int j = 0; j < 4; ++j) {
            int n = nbase + nsub * 64 + j * 16 + l16;
            int comp = n >> 10, rem = n & 1023, h = rem >> 6, dh = rem & 63;
            float bv = bias[n];
            float* dst = qkvf + ((((size_t)comp * B_ + b_) * H_ + h) * DH_ + dh) * S_ + s0;
#pragma unroll
            for (int r = 0; r < 8; ++r) dst[r] = acc[i][j][r] + bv;
        }
    }
}

// ---------------- K2: L2-normalize Q,K rows along S, emit bf16 -------------------

__global__ void k_norm_rows(const float* __restrict__ qkvf,
                            __bf16* __restrict__ Qb, __bf16* __restrict__ Kb) {
    int rid  = blockIdx.x;                // 0..8191 (Q rows then K rows)
    int comp = rid >> 12;
    int row  = rid & 4095;
    const float* src = qkvf + (size_t)rid * S_;
    __bf16* dst = (comp ? Kb : Qb) + (size_t)row * S_;

    __shared__ float red[256];
    float ss = 0.f;
    for (int s = threadIdx.x; s < S_; s += 256) { float v = src[s]; ss += v * v; }
    red[threadIdx.x] = ss;
    __syncthreads();
    for (int o = 128; o > 0; o >>= 1) {
        if ((int)threadIdx.x < o) red[threadIdx.x] += red[threadIdx.x + o];
        __syncthreads();
    }
    float inv = 1.0f / fmaxf(sqrtf(red[0]), 1e-12f);
    for (int s = threadIdx.x; s < S_; s += 256) dst[s] = (__bf16)(src[s] * inv);
}

// ---------------- K3: V -> Vt bf16 [B,H,S,Dh] (transposed for B-fragments) -------

__global__ void k_v_transpose(const float* __restrict__ qkvf, __bf16* __restrict__ Vt) {
    int g  = blockIdx.x;                 // B*H*(S/64) = 4096
    int bh = g >> 6;
    int st = g & 63;
    const float* src = qkvf + ((size_t)2 * B_ * H_ * DH_ + (size_t)bh * DH_) * S_ + st * 64;
    __shared__ float tile[64][65];
    for (int i = threadIdx.x; i < 4096; i += 256) {
        int dh = i >> 6, s = i & 63;
        tile[dh][s] = src[(size_t)dh * S_ + s];
    }
    __syncthreads();
    __bf16* dst = Vt + ((size_t)bh * S_ + st * 64) * DH_;
    for (int i = threadIdx.x; i < 4096; i += 256) {
        int s = i >> 6, dh = i & 63;
        dst[(size_t)s * DH_ + dh] = (__bf16)tile[dh][s];
    }
}

// ---------------- K4: attention core per (b,h) -----------------------------------
// w = (Qn Kn^T / 32) * temp[h]; softmax rows; out = attn x V -> Ob bf16 [B,S,HID]

__global__ void k_xca_attn(const __bf16* __restrict__ Qb, const __bf16* __restrict__ Kb,
                           const __bf16* __restrict__ Vt, const float* __restrict__ temperature,
                           __bf16* __restrict__ Ob) {
    int bh = blockIdx.x;                  // b*H + h
    int b  = bh >> 4, h = bh & 15;
    int tid = threadIdx.x, wave = tid >> 5, lane = tid & 31;
    int l16 = lane & 15, hf = lane >> 4;

    __shared__ float  wmat[64][64];
    __shared__ __bf16 attnb[64][64];

    const __bf16* Q = Qb + (size_t)bh * DH_ * S_;
    const __bf16* K = Kb + (size_t)bh * DH_ * S_;
    const __bf16* V = Vt + (size_t)bh * S_ * DH_;
    float tscale = temperature[h] * (1.0f / 32.0f);   // 1/sqrt(HID) = 1/32

    // Phase A: 64x64 scores = 16 tiles (4x4); 2 tiles per wave, reduce over S.
    for (int t = wave * 2; t < wave * 2 + 2; ++t) {
        int dmt = t >> 2, emt = t & 3;
        const __bf16* Abase = Q + (size_t)dmt * 16 * S_;
        const __bf16* Bbase = K + (size_t)emt * 16 * S_;
        v8f acc = {};
        for (int s0 = 0; s0 < S_; s0 += 32) {
            __builtin_prefetch(Abase + (size_t)l16 * S_ + s0 + 256, 0, 1);
            __builtin_prefetch(Bbase + (size_t)l16 * S_ + s0 + 256, 0, 1);
            v16bf a  = load_frag_a(Abase + s0, S_, l16, hf);
            v16bf bf = load_frag_b(Bbase + s0, S_, l16, hf);
            acc = wmma_bf16(a, bf, acc);
        }
        int e  = emt * 16 + l16;
        int d0 = dmt * 16 + hf * 8;
#pragma unroll
        for (int r = 0; r < 8; ++r) wmat[d0 + r][e] = acc[r] * tscale;
    }
    __syncthreads();

    // Phase B: softmax over rows (64 elems = 2 per lane), wave32 shuffles.
    for (int r = wave * 8; r < wave * 8 + 8; ++r) {
        float x0 = wmat[r][lane], x1 = wmat[r][lane + 32];
        float mx = fmaxf(x0, x1);
        for (int off = 16; off > 0; off >>= 1) mx = fmaxf(mx, __shfl_xor(mx, off, 32));
        float e0 = __expf(x0 - mx), e1 = __expf(x1 - mx);
        float sm = e0 + e1;
        for (int off = 16; off > 0; off >>= 1) sm += __shfl_xor(sm, off, 32);
        float inv = 1.0f / sm;
        attnb[r][lane]      = (__bf16)(e0 * inv);
        attnb[r][lane + 32] = (__bf16)(e1 * inv);
    }
    __syncthreads();

    // Phase C: out[d][s] = sum_e attn[d][e] * V[e][s];  A from LDS, B from Vt.
    int dmt = wave & 3;
    int ng  = wave >> 2;
    v16bf a0 = load_frag_a(&attnb[dmt * 16][0],  64, l16, hf);
    v16bf a1 = load_frag_a(&attnb[dmt * 16][32], 64, l16, hf);
    for (int nt = ng; nt < S_ / 16; nt += 2) {
        int sb = nt * 16;
        const __bf16* Vb = V + (size_t)sb * DH_;
        v8f acc = {};
        v16bf b0 = load_frag_b(Vb,      DH_, l16, hf);   // e = 0..31
        v16bf b1 = load_frag_b(Vb + 32, DH_, l16, hf);   // e = 32..63
        acc = wmma_bf16(a0, b0, acc);
        acc = wmma_bf16(a1, b1, acc);
        int s  = sb + l16;
        int d0 = dmt * 16 + hf * 8;
        v8bf o;
#pragma unroll
        for (int r = 0; r < 8; ++r) o[r] = (__bf16)acc[r];
        __bf16* dst = Ob + ((size_t)b * S_ + s) * HID_ + h * DH_ + d0;
        *(v8bf*)dst = o;
    }
}

// ---------------- K5: output projection GEMM, TDM-staged -> d_out fp32 -----------

__global__ void k_out_gemm(const __bf16* __restrict__ A, const __bf16* __restrict__ W,
                           const float* __restrict__ bias, float* __restrict__ out) {
    __shared__ __bf16 As[128][LDS_STRIDE];
    __shared__ __bf16 Bs[128][LDS_STRIDE];

    int bm = blockIdx.x & 127;        // 128 M-blocks
    int bn = blockIdx.x >> 7;         // 8   N-blocks
    int mbase = bm * 128, nbase = bn * 128;
    int wave = threadIdx.x >> 5, lane = threadIdx.x & 31;
    int l16 = lane & 15, hf = lane >> 4;
    int msub = wave & 3;
    int nsub = wave >> 2;

    unsigned ldsA = lds_offset_of(&As[0][0]);
    unsigned ldsB = lds_offset_of(&Bs[0][0]);

    v8f acc[2][4] = {};
    for (int k0 = 0; k0 < HID_; k0 += 32) {
        if (wave == 0) {
            tdm_load_tile_2d(A + (size_t)mbase * HID_ + k0, ldsA,
                             HID_, 16384u, 32u, 128u, HID_);
            tdm_load_tile_2d(W + (size_t)nbase * HID_ + k0, ldsB,
                             HID_, 1024u, 32u, 128u, HID_);
            __builtin_amdgcn_s_wait_tensorcnt((short)0);
        }
        __syncthreads();
        v16bf a0 = load_frag_a(&As[msub * 32][0],      LDS_STRIDE, l16, hf);
        v16bf a1 = load_frag_a(&As[msub * 32 + 16][0], LDS_STRIDE, l16, hf);
#pragma unroll
        for (int j = 0; j < 4; ++j) {
            v16bf bf = load_frag_b(&Bs[nsub * 64 + j * 16][0], LDS_STRIDE, l16, hf);
            acc[0][j] = wmma_bf16(a0, bf, acc[0][j]);
            acc[1][j] = wmma_bf16(a1, bf, acc[1][j]);
        }
        __syncthreads();
    }

#pragma unroll
    for (int i = 0; i < 2; ++i) {
        int m0 = mbase + msub * 32 + i * 16 + hf * 8;
#pragma unroll
        for (int j = 0; j < 4; ++j) {
            int n = nbase + nsub * 64 + j * 16 + l16;
            float bv = bias[n];
#pragma unroll
            for (int r = 0; r < 8; ++r) out[(size_t)(m0 + r) * HID_ + n] = acc[i][j][r] + bv;
        }
    }
}

// ---------------- host: launch pipeline ------------------------------------------

extern "C" void kernel_launch(void* const* d_in, const int* in_sizes, int n_in,
                              void* d_out, int out_size, void* d_ws, size_t ws_size,
                              hipStream_t stream) {
    const float* x    = (const float*)d_in[0];   // [B,S,HID]
    const float* Wqkv = (const float*)d_in[1];   // [3HID,HID]
    const float* bqkv = (const float*)d_in[2];   // [3HID]
    const float* Wz   = (const float*)d_in[3];   // [HID,HID]
    const float* bz   = (const float*)d_in[4];   // [HID]
    const float* temp = (const float*)d_in[5];   // [H]
    float* out = (float*)d_out;

    const size_t NX = (size_t)B_ * S_ * HID_;
    const size_t NW = (size_t)3 * HID_ * HID_;
    const size_t NZ = (size_t)HID_ * HID_;

    size_t off = 0;
    auto carve = [&](size_t bytes) {
        char* p = (char*)d_ws + off;
        off += (bytes + 255) & ~(size_t)255;
        return (void*)p;
    };
    __bf16* Xb   = (__bf16*)carve(NX * 2);
    __bf16* Wqb  = (__bf16*)carve(NW * 2);
    __bf16* Wzb  = (__bf16*)carve(NZ * 2);
    float*  qkvf = (float*)carve(3 * NX * 4);        // [3,B,H,Dh,S]
    __bf16* Qb   = (__bf16*)carve(NX * 2);           // [B,H,Dh,S] normalized
    __bf16* Kb   = (__bf16*)carve(NX * 2);
    __bf16* Vt   = (__bf16*)carve(NX * 2);           // [B,H,S,Dh]
    __bf16* Ob   = (__bf16*)carve(NX * 2);           // [B,S,HID]
    (void)ws_size; (void)in_sizes; (void)n_in; (void)out_size;

    k_to_bf16<<<(int)((NX + 255) / 256), 256, 0, stream>>>(x, Xb, (int)NX);
    k_to_bf16<<<(int)((NW + 255) / 256), 256, 0, stream>>>(Wqkv, Wqb, (int)NW);
    k_to_bf16<<<(int)((NZ + 255) / 256), 256, 0, stream>>>(Wz, Wzb, (int)NZ);

    // K1: QKV GEMM: (16384/128) x (3072/128) = 128 x 24 blocks
    k_qkv_gemm<<<128 * 24, 256, 0, stream>>>(Xb, Wqb, bqkv, qkvf);

    // K2: normalize Q,K rows
    k_norm_rows<<<2 * B_ * H_ * DH_, 256, 0, stream>>>(qkvf, Qb, Kb);

    // K3: V transpose -> bf16
    k_v_transpose<<<B_ * H_ * (S_ / 64), 256, 0, stream>>>(qkvf, Vt);

    // K4: attention per (b,h)
    k_xca_attn<<<B_ * H_, 256, 0, stream>>>(Qb, Kb, Vt, temp, Ob);

    // K5: output projection: 128 x 8 blocks
    k_out_gemm<<<128 * 8, 256, 0, stream>>>(Ob, Wzb, bz, out);
}
